// PUNet_1932735283294
// MI455X (gfx1250) — matrix operand order repaired and checked by reading, not compiled, not run on hardware
//
#include <hip/hip_runtime.h>

// ---------------------------------------------------------------------------
// PUNet-style GNN for MI455X (gfx1250, wave32, WMMA f16 16x16x32, f32 accum)
// Heavy GEMMs (graph conv over 2*E=524288 gathered rows, final edge head,
// edge-MLP) run on v_wmma_f32_16x16x32_f16. Big weight blocks are staged into
// LDS by the Tensor Data Mover (tensor_load_to_lds + s_wait_tensorcnt).
// Scatter-add uses inline-asm global_atomic_add_f32.
// ---------------------------------------------------------------------------

#define NN   2048
#define EE   262144

typedef __attribute__((ext_vector_type(16))) _Float16 v16h;
typedef __attribute__((ext_vector_type(8)))  _Float16 v8h;
typedef __attribute__((ext_vector_type(8)))  float    v8f;
typedef unsigned int u32x4 __attribute__((ext_vector_type(4)));
typedef int          i32x8 __attribute__((ext_vector_type(8)));
typedef int          i32x4 __attribute__((ext_vector_type(4)));

__device__ __forceinline__ float lrelu(float x) { return x > 0.f ? x : 0.01f * x; }
__device__ __forceinline__ float sigm(float x)  { return 1.f / (1.f + __expf(-x)); }

__device__ __forceinline__ void atomAddF(float* p, float v) {
  asm volatile("global_atomic_add_f32 %0, %1, off" :: "v"(p), "v"(v) : "memory");
}

// TDM: linear copy global -> LDS, expressed as one 2D tile of 8-byte elements.
// D# layout per CDNA5 ISA ch.8 (group0: count/lds_addr/global_addr/type,
// group1: data_size, tensor_dim0/1, tile_dim0/1, stride0).
__device__ __forceinline__ void tdm_copy_to_lds(void* lds_dst, const void* gsrc,
                                                unsigned bytes) {
  unsigned lo = (unsigned)(unsigned long long)(size_t)lds_dst;  // LDS byte offset
  unsigned long long ga = (unsigned long long)(size_t)gsrc;
  unsigned ndw = bytes >> 3;                                    // 8B elements
  u32x4 g0;
  g0[0] = 1u;                                                   // count=1 (valid)
  g0[1] = lo;                                                   // lds_addr
  g0[2] = (unsigned)(ga & 0xffffffffu);                         // global_addr lo
  g0[3] = (unsigned)((ga >> 32) & 0x1ffffffu) | (2u << 30);     // addr hi | type=2
  i32x8 g1;
  g1[0] = (int)(3u << 16);                                      // data_size = 8B
  g1[1] = (int)((ndw & 0xffffu) << 16);                         // tensor_dim0 lo16
  g1[2] = (int)((ndw >> 16) | (1u << 16));                      // dim0 hi | dim1=1
  g1[3] = (int)((ndw & 0xffffu) << 16);                         // tile_dim0
  g1[4] = 1;                                                    // tile_dim1=1
  g1[5] = (int)ndw;                                             // dim0_stride lo
  g1[6] = 0;
  g1[7] = 0;
  i32x4 z4 = {0, 0, 0, 0};
  i32x8 z8 = {0, 0, 0, 0, 0, 0, 0, 0};
  __builtin_amdgcn_tensor_load_to_lds(g0, g1, z4, z4, z8, 0);
}

// A fragment (16x32, MxK), row-major, lda in halfs.
// lane: m = lane&15, g = lane>>4 ; element e -> k = e + 8g + (e>=8 ? 8 : 0)
__device__ __forceinline__ v16h frag_a(const _Float16* base, int lda, int lane) {
  const _Float16* p = base + (lane & 15) * lda + ((lane >> 4) << 3);
  v8h lo = *(const v8h*)p;
  v8h hi = *(const v8h*)(p + 16);
  v16h r;
#pragma unroll
  for (int i = 0; i < 8; ++i) { r[i] = lo[i]; r[i + 8] = hi[i]; }
  return r;
}

// B fragment (32x16, KxN) from TRANSPOSED weights Wt[N][K], ldk in halfs.
// lane: n = lane&15, g = lane>>4 ; element e -> k = e + 8g + (e>=8 ? 8 : 0)
__device__ __forceinline__ v16h frag_bt(const _Float16* baseT, int ldk, int lane) {
  const _Float16* p = baseT + (lane & 15) * ldk + ((lane >> 4) << 3);
  v8h lo = *(const v8h*)p;
  v8h hi = *(const v8h*)(p + 16);
  v16h r;
#pragma unroll
  for (int i = 0; i < 8; ++i) { r[i] = lo[i]; r[i + 8] = hi[i]; }
  return r;
}

// 64x64 @ 64x64 (K=64) tile GEMM + bias + leaky -> LDS. mt fixed per wave.
__device__ __forceinline__ void gemm64_tile(const _Float16* A, const _Float16* Wt,
                                            const float* bias, _Float16* OutLds,
                                            int lane, int wv) {
  int mt = wv & 3, nb = wv >> 2;
  v16h a0 = frag_a(A + mt * 16 * 64, 64, lane);
  v16h a1 = frag_a(A + mt * 16 * 64 + 32, 64, lane);
#pragma unroll
  for (int j = 0; j < 2; ++j) {
    int nt = nb + 2 * j;
    v8f c = {};
    c = __builtin_amdgcn_wmma_f32_16x16x32_f16(false, a0, false,
          frag_bt(Wt + nt * 16 * 64, 64, lane), (short)0, c, false, false);
    c = __builtin_amdgcn_wmma_f32_16x16x32_f16(false, a1, false,
          frag_bt(Wt + nt * 16 * 64 + 32, 64, lane), (short)0, c, false, false);
    int nl = nt * 16 + (lane & 15);
    int mb = mt * 16 + ((lane >> 4) << 3);
    float bv = bias[nl];
#pragma unroll
    for (int jj = 0; jj < 8; ++jj) {
      float v = c[jj] + bv;
      OutLds[(mb + jj) * 64 + nl] = (_Float16)(v > 0.f ? v : 0.01f * v);
    }
  }
}

// --------------------------- weight prep (f16, transposed) ------------------
__global__ void k_prep(const float* __restrict__ gc_w1, const float* __restrict__ gc_w2,
                       const float* __restrict__ en_w1,
                       _Float16* __restrict__ w1t, _Float16* __restrict__ w2t,
                       _Float16* __restrict__ ent) {
  int idx = blockIdx.x * blockDim.x + threadIdx.x;
  if (idx >= 138240) return;
  if (idx < 73728) {                       // gc_w1: 2 x [192n][192k]
    int l = idx / 36864, r = idx % 36864;
    int n = r / 192, k = r % 192;
    w1t[idx] = (_Float16)gc_w1[l * 36864 + k * 192 + n];
  } else if (idx < 110592) {               // gc_w2: 2 x [96n][192k]
    int j = idx - 73728;
    int l = j / 18432, r = j % 18432;
    int n = r / 192, k = r % 192;
    w2t[j] = (_Float16)gc_w2[l * 18432 + k * 96 + n];
  } else {                                 // en_w1: [96n][288k], k>=268 zero pad
    int j = idx - 110592;
    int n = j / 288, k = j % 288;
    ent[j] = (k < 268) ? (_Float16)en_w1[k * 96 + n] : (_Float16)0.f;
  }
}

// --------------------------- edge MLP + gates (WMMA) ------------------------
__global__ void k_edge_mlp(const float* __restrict__ efeat, const float* __restrict__ escal,
                           const float* ein_w1, const float* ein_b1,
                           const float* ein_w2, const float* ein_b2,
                           const float* ad_w1, const float* ad_b1,
                           const float* ad_w2, const float* ad_b2,
                           const float* ar_w1, const float* ar_b1,
                           const float* ar_w2, const float* ar_b2,
                           _Float16* __restrict__ efnn16, float* __restrict__ alpha, int E) {
  __shared__ float W1in[12 * 64];
  __shared__ float b1in[64];
  __shared__ __align__(16) _Float16 W2t[64 * 64];
  __shared__ __align__(16) _Float16 WDt[64 * 64];
  __shared__ __align__(16) _Float16 WRt[64 * 64];
  __shared__ float b2s[64], bDs[64], wD2[64], bRs[64], wR2[64];
  __shared__ float efs[64 * 12];
  __shared__ __align__(16) _Float16 H1[64 * 64];
  __shared__ __align__(16) _Float16 H2[64 * 64];
  __shared__ __align__(16) _Float16 H3[64 * 64];
  __shared__ float esc[12];
  __shared__ float c2[2];
  int tid = threadIdx.x, lane = tid & 31, wv = tid >> 5;
  for (int i = tid; i < 12 * 64; i += 256) W1in[i] = ein_w1[i];
  for (int i = tid; i < 64 * 64; i += 256) {
    int n = i >> 6, k = i & 63;
    W2t[i] = (_Float16)ein_w2[k * 64 + n];
    WDt[i] = (_Float16)ad_w1[k * 64 + n];
    WRt[i] = (_Float16)ar_w1[k * 64 + n];
  }
  if (tid < 64) {
    b1in[tid] = ein_b1[tid]; b2s[tid] = ein_b2[tid];
    bDs[tid] = ad_b1[tid];   wD2[tid] = ad_w2[tid];
    bRs[tid] = ar_b1[tid];   wR2[tid] = ar_w2[tid];
  }
  if (tid < 12) esc[tid] = escal[tid];
  if (tid == 0) { c2[0] = ad_b2[0]; c2[1] = ar_b2[0]; }
  __syncthreads();
  int ntiles = E >> 6;
  for (int t = blockIdx.x; t < ntiles; t += gridDim.x) {
    for (int i = tid; i < 64 * 12; i += 256) {
      int row = i / 12, j = i - row * 12;
      efs[i] = (efeat[(size_t)((t << 6) + row) * 12 + j] + 1e-4f) / esc[j];
    }
    __syncthreads();
    {   // layer 1 (K=12) on VALU
      int row = tid >> 2, q = tid & 3;
      const float* er = efs + row * 12;
      for (int c = (q << 4); c < (q << 4) + 16; ++c) {
        float s = b1in[c];
#pragma unroll
        for (int j = 0; j < 12; ++j) s += er[j] * W1in[j * 64 + c];
        H1[row * 64 + c] = (_Float16)lrelu(s);
      }
    }
    __syncthreads();
    gemm64_tile(H1, W2t, b2s, H2, lane, wv);   // ef_nn tile
    __syncthreads();
    {   // coalesced H2 -> global (tile rows contiguous in efnn16)
      uint4* g = (uint4*)(efnn16 + (size_t)(t << 6) * 64);
      const uint4* s = (const uint4*)H2;
#pragma unroll
      for (int i = tid; i < 512; i += 256) g[i] = s[i];
    }
    gemm64_tile(H2, WDt, bDs, H3, lane, wv);
    __syncthreads();
    if (tid < 64) {
      float s = c2[0];
#pragma unroll
      for (int c = 0; c < 64; ++c) s += (float)H3[tid * 64 + c] * wD2[c];
      alpha[(t << 6) + tid] = sigm(s);
    }
    __syncthreads();
    gemm64_tile(H2, WRt, bRs, H3, lane, wv);
    __syncthreads();
    if (tid < 64) {
      float s = c2[1];
#pragma unroll
      for (int c = 0; c < 64; ++c) s += (float)H3[tid * 64 + c] * wR2[c];
      alpha[E + (t << 6) + tid] = sigm(s);
    }
    __syncthreads();
  }
}

// --------------------------- small node-side kernels ------------------------
__global__ void k_scaleX(const float* __restrict__ X, const float* __restrict__ ns,
                         float* __restrict__ out, int total) {
  int idx = blockIdx.x * blockDim.x + threadIdx.x;
  if (idx < total) out[idx] = X[idx] / ns[idx % 19];
}

__global__ void k_mm(const float* __restrict__ A, const float* __restrict__ W,
                     const float* __restrict__ b, float* __restrict__ out,
                     int M, int K, int Nc, int act) {
  int idx = blockIdx.x * blockDim.x + threadIdx.x;
  int total = M * Nc;
  if (idx >= total) return;
  int row = idx / Nc, col = idx - row * Nc;
  float s = b[col];
  const float* a = A + (size_t)row * K;
  for (int k = 0; k < K; ++k) s += a[k] * W[(size_t)k * Nc + col];
  if (act == 1) s = lrelu(s);
  else if (act == 2) s = fmaxf(s, 0.f);
  out[idx] = s;
}

__global__ void k_attn(const float* __restrict__ qkv, float* __restrict__ o) {
  __shared__ float Kh[2048 * 8];
  __shared__ float Vh[2048 * 8];
  int head = blockIdx.x, chunk = blockIdx.y, tid = threadIdx.x;
  for (int i = tid; i < 2048 * 8; i += 256) {
    int m = i >> 3, d = i & 7;
    Kh[i] = qkv[(size_t)m * 288 + 96 + head * 8 + d];
    Vh[i] = qkv[(size_t)m * 288 + 192 + head * 8 + d];
  }
  __syncthreads();
  int row = chunk * 256 + tid;
  float q[8];
#pragma unroll
  for (int d = 0; d < 8; ++d)
    q[d] = qkv[(size_t)row * 288 + head * 8 + d] * 0.35355339059327373f;  // 1/sqrt(8)
  float mmax = -1e30f, l = 0.f, acc[8] = {0, 0, 0, 0, 0, 0, 0, 0};
  for (int m = 0; m < 2048; ++m) {
    float s = 0.f;
#pragma unroll
    for (int d = 0; d < 8; ++d) s += q[d] * Kh[m * 8 + d];
    float nm = fmaxf(mmax, s);
    float corr = __expf(mmax - nm);
    float p = __expf(s - nm);
    l = l * corr + p;
#pragma unroll
    for (int d = 0; d < 8; ++d) acc[d] = acc[d] * corr + p * Vh[m * 8 + d];
    mmax = nm;
  }
  float inv = 1.f / l;
#pragma unroll
  for (int d = 0; d < 8; ++d) o[(size_t)row * 96 + head * 8 + d] = acc[d] * inv;
}

__global__ void k_add_ln(float* __restrict__ h, const float* __restrict__ r,
                         const float* __restrict__ s, const float* __restrict__ b, int M) {
  int row = blockIdx.x * blockDim.x + threadIdx.x;
  if (row >= M) return;
  const float* hp = h + (size_t)row * 96;
  const float* rp = r + (size_t)row * 96;
  float mean = 0.f;
  for (int c = 0; c < 96; ++c) mean += hp[c] + rp[c];
  mean *= (1.f / 96.f);
  float var = 0.f;
  for (int c = 0; c < 96; ++c) { float d = hp[c] + rp[c] - mean; var += d * d; }
  var *= (1.f / 96.f);
  float inv = rsqrtf(var + 1e-5f);
  for (int c = 0; c < 96; ++c)
    h[(size_t)row * 96 + c] = (hp[c] + rp[c] - mean) * inv * s[c] + b[c];
}

__global__ void k_cvt(const float* __restrict__ a, _Float16* __restrict__ b, int n) {
  int idx = blockIdx.x * blockDim.x + threadIdx.x;
  if (idx < n) b[idx] = (_Float16)a[idx];
}

__global__ void k_zero(float* __restrict__ p, int n) {
  int idx = blockIdx.x * blockDim.x + threadIdx.x;
  if (idx < n) p[idx] = 0.f;
}

// den[seg] = sum alpha_full over ind1 (iteration-invariant); first N rows -> seg 0
__global__ void k_den(const int* __restrict__ ei, const float* __restrict__ alpha,
                      float* __restrict__ den, int E, int Nn) {
  int idx = blockIdx.x * blockDim.x + threadIdx.x;
  if (idx == 0) atomAddF(&den[0], (float)Nn);
  for (int r = idx; r < 2 * E; r += gridDim.x * blockDim.x) {
    int e = (r < E) ? r : r - E;
    int seg = (r < E) ? ei[2 * e] : ei[2 * e + 1];
    atomAddF(&den[seg], alpha[r]);
  }
}

// num[0] += N * MLP(concat(h[0],h[0]))  (the N identical zero-index rows)
__global__ void k_seed(const float* __restrict__ h, const float* __restrict__ w1,
                       const float* __restrict__ b1, const float* __restrict__ w2,
                       const float* __restrict__ b2, float* __restrict__ num, int Nn) {
  __shared__ float hrow[192];
  __shared__ float hid[192];
  int tid = threadIdx.x;
  if (tid < 96) { hrow[tid] = h[tid]; hrow[96 + tid] = h[tid]; }
  __syncthreads();
  if (tid < 192) {
    float s = b1[tid];
    for (int k = 0; k < 192; ++k) s += hrow[k] * w1[k * 192 + tid];
    hid[tid] = lrelu(s);
  }
  __syncthreads();
  if (tid < 96) {
    float s = b2[tid];
    for (int k = 0; k < 192; ++k) s += hid[k] * w2[k * 96 + tid];
    num[tid] += (float)Nn * lrelu(s);
  }
}

// --------------------------- graph conv (dominant, WMMA) --------------------
__global__ void k_gconv(const _Float16* __restrict__ h16, const int* __restrict__ ei,
                        const float* __restrict__ alpha,
                        const _Float16* __restrict__ w1t, const float* __restrict__ b1,
                        const _Float16* __restrict__ w2t, const float* __restrict__ b2,
                        float* __restrict__ num, int E) {
  __shared__ __align__(16) _Float16 W1s[192 * 192];   // transposed [192n][192k]
  __shared__ __align__(16) _Float16 W2s[96 * 192];    // transposed [96n][192k]
  __shared__ __align__(16) _Float16 As[64 * 192];
  __shared__ __align__(16) _Float16 Bs[64 * 192];
  __shared__ float b1s[192];
  __shared__ float b2s[96];
  __shared__ int   segs[64];
  __shared__ float als[64];
  int tid = threadIdx.x, lane = tid & 31, wv = tid >> 5;
  if (wv == 0) {     // Tensor Data Mover: DMA both weight blocks into LDS
    tdm_copy_to_lds(W1s, w1t, 192 * 192 * 2);
    tdm_copy_to_lds(W2s, w2t, 96 * 192 * 2);
    __builtin_amdgcn_s_wait_tensorcnt(0);
  }
  if (tid < 192) b1s[tid] = b1[tid];
  if (tid < 96)  b2s[tid] = b2[tid];
  __syncthreads();
  int mt = wv & 3, nb = wv >> 2;
  int ntiles = (2 * E) >> 6;
  for (int t = blockIdx.x; t < ntiles; t += gridDim.x) {
    {   // gather concat(h16[ind1], h16[ind2]) -> As (64 x 192 f16)
      int row = tid >> 2, sub = tid & 3;
      int r = (t << 6) + row;
      int e = (r < E) ? r : r - E;
      int s0 = ei[2 * e], s1 = ei[2 * e + 1];
      int n1 = (r < E) ? s0 : s1;
      int n2 = (r < E) ? s1 : s0;
      if (sub == 0) { segs[row] = n1; als[row] = alpha[r]; }
      int node = (sub < 2) ? n1 : n2;
      const uint4* sp = (const uint4*)(h16 + (size_t)node * 96 + (sub & 1) * 48);
      uint4* dp = (uint4*)(As + row * 192 + sub * 48);
      dp[0] = sp[0]; dp[1] = sp[1]; dp[2] = sp[2];
    }
    __syncthreads();
    // GEMM1: (64x192) x (192x192), leaky -> Bs. A fragments hoisted (mt fixed).
    {
      v16h a[6];
#pragma unroll
      for (int kt = 0; kt < 6; ++kt)
        a[kt] = frag_a(As + mt * 16 * 192 + kt * 32, 192, lane);
#pragma unroll
      for (int j = 0; j < 6; ++j) {
        int nt = nb + 2 * j;
        v8f c = {};
#pragma unroll
        for (int kt = 0; kt < 6; ++kt)
          c = __builtin_amdgcn_wmma_f32_16x16x32_f16(false, a[kt], false,
                frag_bt(W1s + nt * 16 * 192 + kt * 32, 192, lane), (short)0, c, false, false);
        int nl = nt * 16 + (lane & 15);
        int mb = mt * 16 + ((lane >> 4) << 3);
        float bv = b1s[nl];
#pragma unroll
        for (int jj = 0; jj < 8; ++jj) {
          float v = c[jj] + bv;
          Bs[(mb + jj) * 192 + nl] = (_Float16)(v > 0.f ? v : 0.01f * v);
        }
      }
    }
    __syncthreads();
    // GEMM2: (64x192) x (192x96), leaky, *alpha, atomic scatter into num[seg]
    {
      v16h a[6];
#pragma unroll
      for (int kt = 0; kt < 6; ++kt)
        a[kt] = frag_a(Bs + mt * 16 * 192 + kt * 32, 192, lane);
#pragma unroll
      for (int j = 0; j < 3; ++j) {
        int nt = nb + 2 * j;
        v8f c = {};
#pragma unroll
        for (int kt = 0; kt < 6; ++kt)
          c = __builtin_amdgcn_wmma_f32_16x16x32_f16(false, a[kt], false,
                frag_bt(W2s + nt * 16 * 192 + kt * 32, 192, lane), (short)0, c, false, false);
        int nl = nt * 16 + (lane & 15);
        int mb = mt * 16 + ((lane >> 4) << 3);
        float bv = b2s[nl];
#pragma unroll
        for (int jj = 0; jj < 8; ++jj) {
          int row = mb + jj;
          float v = c[jj] + bv;
          v = (v > 0.f ? v : 0.01f * v) * als[row];
          atomAddF(num + (size_t)segs[row] * 96 + nl, v);
        }
      }
    }
    __syncthreads();
  }
}

__global__ void k_hupdate(const float* __restrict__ num, const float* __restrict__ den,
                          float* __restrict__ h, _Float16* __restrict__ h16, int total) {
  int idx = blockIdx.x * blockDim.x + threadIdx.x;
  if (idx >= total) return;
  float v = num[idx] / (den[idx / 96] + 1e-8f);
  h[idx] = v;
  h16[idx] = (_Float16)v;
}

// --------------------------- final edge head (WMMA) -------------------------
__global__ void k_final(const _Float16* __restrict__ h16, const _Float16* __restrict__ efnn16,
                        const float* __restrict__ efeat, const float* __restrict__ escal,
                        const int* __restrict__ ei, const _Float16* __restrict__ w1t,
                        const float* __restrict__ b1, const float* __restrict__ w2,
                        const float* __restrict__ b2, float* __restrict__ out, int E) {
  __shared__ __align__(16) _Float16 Wf[96 * 288];     // transposed [96n][288k]
  __shared__ __align__(16) _Float16 As[64 * 288];
  __shared__ __align__(16) _Float16 Hs[64 * 96];
  __shared__ float b1s[96], w2s[96], esc[12], sb2[1];
  int tid = threadIdx.x, lane = tid & 31, wv = tid >> 5;
  if (wv == 0) {     // TDM stage of padded en_w1 (transposed)
    tdm_copy_to_lds(Wf, w1t, 96 * 288 * 2);
    __builtin_amdgcn_s_wait_tensorcnt(0);
  }
  if (tid < 96) { b1s[tid] = b1[tid]; w2s[tid] = w2[tid]; }
  if (tid < 12) esc[tid] = escal[tid];
  if (tid == 0) sb2[0] = b2[0];
  __syncthreads();
  int mt = wv & 3, nb = wv >> 2;
  int ntiles = E >> 6;
  for (int t = blockIdx.x; t < ntiles; t += gridDim.x) {
    {   // vectorized gather: [h(src) | h(dst) | ef_nn | ef | pad] per row
      int row = tid >> 2, sub = tid & 3;
      int e = (t << 6) + row;
      if (sub == 0) {
        const uint4* sp = (const uint4*)(h16 + (size_t)ei[2 * e] * 96);
        uint4* dp = (uint4*)(As + row * 288);
#pragma unroll
        for (int i = 0; i < 6; ++i) dp[i] = sp[i];
      } else if (sub == 1) {
        const uint4* sp = (const uint4*)(h16 + (size_t)ei[2 * e + 1] * 96);
        uint4* dp = (uint4*)(As + row * 288 + 96);
#pragma unroll
        for (int i = 0; i < 6; ++i) dp[i] = sp[i];
      } else if (sub == 2) {
        const uint4* sp = (const uint4*)(efnn16 + (size_t)e * 64);
        uint4* dp = (uint4*)(As + row * 288 + 192);
#pragma unroll
        for (int i = 0; i < 8; ++i) dp[i] = sp[i];
      } else {
        _Float16* d = As + row * 288 + 256;
#pragma unroll
        for (int j = 0; j < 12; ++j)
          d[j] = (_Float16)((efeat[(size_t)e * 12 + j] + 1e-4f) / esc[j]);
#pragma unroll
        for (int j = 12; j < 32; ++j) d[j] = (_Float16)0.f;
      }
    }
    __syncthreads();
    {
      v16h a[9];
#pragma unroll
      for (int kt = 0; kt < 9; ++kt)
        a[kt] = frag_a(As + mt * 16 * 288 + kt * 32, 288, lane);
#pragma unroll
      for (int j = 0; j < 3; ++j) {
        int nt = nb + 2 * j;
        v8f c = {};
#pragma unroll
        for (int kt = 0; kt < 9; ++kt)
          c = __builtin_amdgcn_wmma_f32_16x16x32_f16(false, a[kt], false,
                frag_bt(Wf + nt * 16 * 288 + kt * 32, 288, lane), (short)0, c, false, false);
        int nl = nt * 16 + (lane & 15);
        int mb = mt * 16 + ((lane >> 4) << 3);
        float bv = b1s[nl];
#pragma unroll
        for (int jj = 0; jj < 8; ++jj) {
          float v = c[jj] + bv;
          Hs[(mb + jj) * 96 + nl] = (_Float16)(v > 0.f ? v : 0.01f * v);
        }
      }
    }
    __syncthreads();
    if (tid < 64) {
      float s = sb2[0];
#pragma unroll
      for (int c = 0; c < 96; ++c) s += (float)Hs[tid * 96 + c] * w2s[c];
      out[(t << 6) + tid] = sigm(s);
    }
    __syncthreads();
  }
}

// --------------------------- host launcher ----------------------------------
extern "C" void kernel_launch(void* const* d_in, const int* in_sizes, int n_in,
                              void* d_out, int out_size, void* d_ws, size_t ws_size,
                              hipStream_t stream) {
  (void)in_sizes; (void)n_in; (void)out_size; (void)ws_size;
  const float* X             = (const float*)d_in[0];
  const float* edge_features = (const float*)d_in[1];
  const int*   edge_index    = (const int*)d_in[2];
  const float* node_scaler   = (const float*)d_in[3];
  const float* edge_scaler   = (const float*)d_in[4];
  const float* in_w1 = (const float*)d_in[5];
  const float* in_b1 = (const float*)d_in[6];
  const float* in_w2 = (const float*)d_in[7];
  const float* in_b2 = (const float*)d_in[8];
  const float* qkv_w = (const float*)d_in[9];
  const float* qkv_b = (const float*)d_in[10];
  const float* out_w = (const float*)d_in[11];
  const float* out_b = (const float*)d_in[12];
  const float* ln1s  = (const float*)d_in[13];
  const float* ln1b  = (const float*)d_in[14];
  const float* ffw1  = (const float*)d_in[15];
  const float* ffb1  = (const float*)d_in[16];
  const float* ffw2  = (const float*)d_in[17];
  const float* ffb2  = (const float*)d_in[18];
  const float* ln2s  = (const float*)d_in[19];
  const float* ln2b  = (const float*)d_in[20];
  const float* ein_w1 = (const float*)d_in[21];
  const float* ein_b1 = (const float*)d_in[22];
  const float* ein_w2 = (const float*)d_in[23];
  const float* ein_b2 = (const float*)d_in[24];
  const float* ad_w1 = (const float*)d_in[25];
  const float* ad_b1 = (const float*)d_in[26];
  const float* ad_w2 = (const float*)d_in[27];
  const float* ad_b2 = (const float*)d_in[28];
  const float* ar_w1 = (const float*)d_in[29];
  const float* ar_b1 = (const float*)d_in[30];
  const float* ar_w2 = (const float*)d_in[31];
  const float* ar_b2 = (const float*)d_in[32];
  const float* gc_w1 = (const float*)d_in[33];
  const float* gc_b1 = (const float*)d_in[34];
  const float* gc_w2 = (const float*)d_in[35];
  const float* gc_b2 = (const float*)d_in[36];
  const float* en_w1 = (const float*)d_in[37];
  const float* en_b1 = (const float*)d_in[38];
  const float* en_w2 = (const float*)d_in[39];
  const float* en_b2 = (const float*)d_in[40];

  char* wp = (char*)d_ws;
  auto carve = [&](size_t bytes) -> void* {
    void* p = (void*)wp; wp += (bytes + 255) & ~(size_t)255; return p;
  };
  _Float16* ef_nn16 = (_Float16*)carve((size_t)EE * 64 * 2);
  float*    alpha   = (float*)carve((size_t)2 * EE * 4);
  float*    hbuf    = (float*)carve((size_t)NN * 96 * 4);
  _Float16* h16     = (_Float16*)carve((size_t)NN * 96 * 2);
  float*    num     = (float*)carve((size_t)NN * 96 * 4);
  float*    den     = (float*)carve((size_t)NN * 4);
  float*    t0      = (float*)carve((size_t)NN * 288 * 4);
  float*    t1      = (float*)carve((size_t)NN * 192 * 4);
  float*    t2      = (float*)carve((size_t)NN * 96 * 4);
  _Float16* w1t     = (_Float16*)carve((size_t)2 * 192 * 192 * 2);
  _Float16* w2t     = (_Float16*)carve((size_t)2 * 96 * 192 * 2);
  _Float16* ent     = (_Float16*)carve((size_t)96 * 288 * 2);

  auto mm = [&](const float* A, const float* W, const float* b, float* o,
                int M, int K, int Nc, int act) {
    int total = M * Nc;
    k_mm<<<dim3((total + 255) / 256), dim3(256), 0, stream>>>(A, W, b, o, M, K, Nc, act);
  };

  // weights -> f16 transposed
  k_prep<<<dim3(540), dim3(256), 0, stream>>>(gc_w1, gc_w2, en_w1, w1t, w2t, ent);

  // edge MLP + gates
  k_edge_mlp<<<dim3(2048), dim3(256), 0, stream>>>(edge_features, edge_scaler,
      ein_w1, ein_b1, ein_w2, ein_b2, ad_w1, ad_b1, ad_w2, ad_b2,
      ar_w1, ar_b1, ar_w2, ar_b2, ef_nn16, alpha, EE);

  // node input MLP
  k_scaleX<<<dim3((NN * 19 + 255) / 256), dim3(256), 0, stream>>>(X, node_scaler, t1, NN * 19);
  mm(t1, in_w1, in_b1, t2, NN, 19, 96, 1);
  mm(t2, in_w2, in_b2, hbuf, NN, 96, 96, 1);

  // 3 encoder layers
  for (int i = 0; i < 3; ++i) {
    mm(hbuf, qkv_w + (size_t)i * 96 * 288, qkv_b + (size_t)i * 288, t0, NN, 96, 288, 0);
    k_attn<<<dim3(12, 8), dim3(256), 0, stream>>>(t0, t1);
    mm(t1, out_w + (size_t)i * 96 * 96, out_b + (size_t)i * 96, t2, NN, 96, 96, 0);
    k_add_ln<<<dim3((NN + 255) / 256), dim3(256), 0, stream>>>(hbuf, t2, ln1s + i * 96, ln1b + i * 96, NN);
    mm(hbuf, ffw1 + (size_t)i * 96 * 192, ffb1 + (size_t)i * 192, t1, NN, 96, 192, 2);
    mm(t1, ffw2 + (size_t)i * 192 * 96, ffb2 + (size_t)i * 96, t2, NN, 192, 96, 0);
    k_add_ln<<<dim3((NN + 255) / 256), dim3(256), 0, stream>>>(hbuf, t2, ln2s + i * 96, ln2b + i * 96, NN);
  }

  // graph conv iterations
  k_cvt<<<dim3((NN * 96 + 255) / 256), dim3(256), 0, stream>>>(hbuf, h16, NN * 96);
  k_zero<<<dim3((NN + 255) / 256), dim3(256), 0, stream>>>(den, NN);
  k_den<<<dim3(2048), dim3(256), 0, stream>>>(edge_index, alpha, den, EE, NN);
  for (int i = 0; i < 2; ++i) {
    k_zero<<<dim3((NN * 96 + 255) / 256), dim3(256), 0, stream>>>(num, NN * 96);
    k_seed<<<dim3(1), dim3(256), 0, stream>>>(hbuf, gc_w1 + (size_t)i * 192 * 192,
        gc_b1 + (size_t)i * 192, gc_w2 + (size_t)i * 192 * 96, gc_b2 + (size_t)i * 96, num, NN);
    k_gconv<<<dim3(1024), dim3(256), 0, stream>>>(h16, edge_index, alpha,
        w1t + (size_t)i * 192 * 192, gc_b1 + (size_t)i * 192,
        w2t + (size_t)i * 96 * 192, gc_b2 + (size_t)i * 96, num, EE);
    k_hupdate<<<dim3((NN * 96 + 255) / 256), dim3(256), 0, stream>>>(num, den, hbuf, h16, NN * 96);
  }

  // final edge head
  k_final<<<dim3(1024), dim3(256), 0, stream>>>(h16, ef_nn16, edge_features, edge_scaler,
      edge_index, ent, en_b1, en_w2, en_b2, (float*)d_out, EE);
}
